// RNN_65352222376427
// MI455X (gfx1250) — matrix operand order repaired and checked by reading, not compiled, not run on hardware
//
#include <hip/hip_runtime.h>
#include <cmath>

#define SEQ_LEN 1024
#define BATCH   1024
#define DIN     64
#define DH      64
#define LDS_STRIDE 72   // bf16 row stride: 36 words/row, 16 rows hit 16 distinct banks

typedef __attribute__((ext_vector_type(16))) __bf16 v16bf;
typedef __attribute__((ext_vector_type(2)))  __bf16 v2bf;
typedef __attribute__((ext_vector_type(8)))  float  v8f;

static __device__ __forceinline__ v8f wmma_bf16(v16bf a, v16bf b, v8f c) {
  // D = A(16x32 bf16) x B(32x16 bf16) + C(16x16 f32); D may differ from C (VOP3P)
  return __builtin_amdgcn_wmma_f32_16x16x32_bf16(
      /*neg_a=*/false, a, /*neg_b=*/false, b,
      /*c_mod=*/(short)0, c, /*reuse_a=*/false, /*reuse_b=*/false);
}

// CDNA5 native TANH transcendental (ISA §8.4) when exposed; branch-free fallback.
static __device__ __forceinline__ float fast_tanh(float x) {
#if __has_builtin(__builtin_amdgcn_tanhf)
  return __builtin_amdgcn_tanhf(x);
#else
  const float e = __builtin_amdgcn_exp2f(x * 2.885390081777927f); // 2*log2(e)
  const float r = __builtin_amdgcn_rcpf(e + 1.0f);
  return __builtin_fmaf(-2.0f, r, 1.0f);
#endif
}

// K offset for element pair p (0..7) of a 16x32 bf16 A fragment
// (lanes 0-15: K in {0..7,16..23}; lanes 16-31: +8) per ISA 7.12.2
static __device__ __forceinline__ int a_koff(int p, int hf) {
  return ((p < 4) ? (2 * p) : (16 + 2 * (p - 4))) + hf * 8;
}

// Build an A fragment (rows = batch, K = feature) from a row-major f32 row.
static __device__ __forceinline__ v16bf loadA_f32(const float* rowbase, int kbase, int hf) {
  v16bf a;
#pragma unroll
  for (int p = 0; p < 8; ++p) {
    const int k = kbase + a_koff(p, hf);
    const float2 v = *(const float2*)(rowbase + k);
    a[2 * p]     = (__bf16)v.x;
    a[2 * p + 1] = (__bf16)v.y;
  }
  return a;
}

// Build an A fragment from a bf16 LDS row (b32 loads, zero conversion ops).
static __device__ __forceinline__ v16bf loadA_bf16(const __bf16* rowbase, int kbase, int hf) {
  v16bf a;
#pragma unroll
  for (int p = 0; p < 8; ++p) {
    const int k = kbase + a_koff(p, hf);
    const v2bf v = *(const v2bf*)(rowbase + k);
    a[2 * p]     = v[0];
    a[2 * p + 1] = v[1];
  }
  return a;
}

__global__ __launch_bounds__(32, 1)
void rnn_scan_fused(const float* __restrict__ x,   // (S,B,DIN)
                    const float* __restrict__ h0,  // (1,B,DH)
                    const float* __restrict__ Wx,  // (DH,DIN) row-major [out,in]
                    const float* __restrict__ bx,  // (DH)
                    const float* __restrict__ Wh,  // (DH,DH)
                    const float* __restrict__ bh,  // (DH)
                    float* __restrict__ out)       // (S,B,DH) then (B,DH)
{
  // h-state staging tile (bf16) + Wx B-fragments parked in LDS (lane-private
  // 32B chunks; contiguous per lane -> conflict-optimal ds_load_b128 pairs).
  __shared__ __bf16 hs[16 * LDS_STRIDE];
  __shared__ __bf16 wx_lds[8 * 32 * 16];   // 8 frags x 32 lanes x 16 bf16 = 8 KB

  const int lane = threadIdx.x;   // wave32
  const int m    = lane & 15;     // row within tile (A-frag row / B-frag col / D col)
  const int hf   = lane >> 4;     // lane half
  const int b0   = blockIdx.x * 16;

  // ---------------- Wh -> B fragments, pinned in VGPRs (on the serial chain).
  // ---------------- Wx -> B fragments, built once then parked in LDS
  //                  (x-projection path is latency-tolerant; frees 64 VGPRs
  //                   so the loop stays under the 256-VGPR direct window).
  v16bf whb[4][2];
#pragma unroll
  for (int nt = 0; nt < 4; ++nt) {
    const float* wxrow = Wx + (size_t)(nt * 16 + m) * DIN;
    const float* whrow = Wh + (size_t)(nt * 16 + m) * DH;
#pragma unroll
    for (int kb = 0; kb < 2; ++kb) {
      v16bf fx, fh;
#pragma unroll
      for (int p = 0; p < 8; ++p) {
        const int k = kb * 32 + a_koff(p, hf);
        const float2 vx = *(const float2*)(wxrow + k);
        const float2 vh = *(const float2*)(whrow + k);
        fx[2 * p] = (__bf16)vx.x; fx[2 * p + 1] = (__bf16)vx.y;
        fh[2 * p] = (__bf16)vh.x; fh[2 * p + 1] = (__bf16)vh.y;
      }
      whb[nt][kb] = fh;
      *(v16bf*)&wx_lds[(((nt * 2 + kb) * 32) + lane) * 16] = fx;
    }
  }

  // ---------------- Bias (bx+bh) broadcast into C/D layout (column = lane&15).
  v8f cinit[4];
#pragma unroll
  for (int nt = 0; nt < 4; ++nt) {
    const float bsum = bx[nt * 16 + m] + bh[nt * 16 + m];
    v8f c;
#pragma unroll
    for (int r = 0; r < 8; ++r) c[r] = bsum;
    cinit[nt] = c;
  }

  // ---------------- Initial hidden state straight from global (row-major f32).
  v16bf ah0 = loadA_f32(h0 + (size_t)(b0 + m) * DH, 0,  hf);
  v16bf ah1 = loadA_f32(h0 + (size_t)(b0 + m) * DH, 32, hf);

  // Per-lane loop-invariant bases (store offsets fold into imm fields).
  const __bf16* wx_base = &wx_lds[(size_t)lane * 16];
  __bf16*       hst     = &hs[hf * 8 * LDS_STRIDE + 0];   // + nn added per nt
  const __bf16* hld     = &hs[m * LDS_STRIDE];

  // ---------------- Prologue: input projection for t=0 (independent of h).
  v8f accx[4];
  {
    const float* xt = x + (size_t)(b0 + m) * DIN;
    const v16bf ax0 = loadA_f32(xt, 0,  hf);
    const v16bf ax1 = loadA_f32(xt, 32, hf);
#pragma unroll
    for (int nt = 0; nt < 4; ++nt) {
      const v16bf w0 = *(const v16bf*)(wx_base + (nt * 2 + 0) * 32 * 16);
      const v16bf w1 = *(const v16bf*)(wx_base + (nt * 2 + 1) * 32 * 16);
      accx[nt] = wmma_bf16(ax1, w1, wmma_bf16(ax0, w0, cinit[nt]));
    }
  }

  for (int t = 0; t < SEQ_LEN; ++t) {
    float* od = out + (size_t)t * BATCH * DH + (size_t)(b0 + hf * 8) * DH;

    // ---- h-dependent part (the true serial chain): 2 chained WMMAs + tanh
#pragma unroll
    for (int nt = 0; nt < 4; ++nt) {
      const v8f acc = wmma_bf16(ah1, whb[nt][1],
                      wmma_bf16(ah0, whb[nt][0], accx[nt]));
      const int nn = nt * 16 + m;              // output column (C/D layout)
      float*   ob  = od + nn;                  // + r*DH imm offsets
      __bf16*  hb  = hst + nn;                 // + r*LDS_STRIDE imm offsets
#pragma unroll
      for (int r = 0; r < 8; ++r) {
        const float hv = fast_tanh(acc[r]);
        ob[r * DH] = hv;                       // h_seq (f32, coalesced across lanes)
        hb[r * LDS_STRIDE] = (__bf16)hv;       // bf16 stage (next step reads bf16 anyway)
      }
    }

    // ---- layout re-shuffle via LDS (in-order within a wave, ds_load_b32 only)
    ah0 = loadA_bf16(hld, 0,  hf);
    ah1 = loadA_bf16(hld, 32, hf);

    // ---- pipeline: next timestep's input projection (independent of h chain)
    if (t + 1 < SEQ_LEN) {
      const float* xn = x + (size_t)(t + 1) * BATCH * DIN + (size_t)(b0 + m) * DIN;
      __builtin_prefetch(xn + (size_t)BATCH * DIN + hf * 32, 0, 3); // warm t+2
      const v16bf ax0 = loadA_f32(xn, 0,  hf);
      const v16bf ax1 = loadA_f32(xn, 32, hf);
#pragma unroll
      for (int nt = 0; nt < 4; ++nt) {
        const v16bf w0 = *(const v16bf*)(wx_base + (nt * 2 + 0) * 32 * 16);
        const v16bf w1 = *(const v16bf*)(wx_base + (nt * 2 + 1) * 32 * 16);
        accx[nt] = wmma_bf16(ax1, w1, wmma_bf16(ax0, w0, cinit[nt]));
      }
    }
  }

  // ---------------- Epilogue: h_last = h_seq[S-1] (tiny global->global copy)
  const float* src = out + (size_t)(SEQ_LEN - 1) * BATCH * DH + (size_t)b0 * DH;
  float*       dst = out + (size_t)SEQ_LEN * BATCH * DH + (size_t)b0 * DH;
  for (int idx = lane; idx < 64; idx += 32) {
    const int row = idx >> 2, c4 = (idx & 3) * 4;
    const float4 v = *(const float4*)(src + (size_t)row * DH + c4);
    *(float4*)(dst + (size_t)row * DH + c4) = v;
  }
}

extern "C" void kernel_launch(void* const* d_in, const int* in_sizes, int n_in,
                              void* d_out, int out_size, void* d_ws, size_t ws_size,
                              hipStream_t stream) {
  (void)in_sizes; (void)n_in; (void)out_size; (void)d_ws; (void)ws_size;
  const float* x  = (const float*)d_in[0];
  const float* h  = (const float*)d_in[1];
  const float* Wx = (const float*)d_in[2];
  const float* bx = (const float*)d_in[3];
  const float* Wh = (const float*)d_in[4];
  const float* bh = (const float*)d_in[5];

  // 64 independent recurrence chains (one 16-row batch tile each), wave32 per WG.
  rnn_scan_fused<<<BATCH / 16, 32, 0, stream>>>(x, h, Wx, bx, Wh, bh, (float*)d_out);
}
